// SpGraphAttentionLayer_4750233829807
// MI455X (gfx1250) — compile-verified, matrix operand order checked by reading, not run on hardware
//
#include <hip/hip_runtime.h>
#include <hip/hip_fp16.h>

typedef _Float16 v16h __attribute__((ext_vector_type(16)));
typedef float    v8f  __attribute__((ext_vector_type(8)));
typedef int      v4i  __attribute__((vector_size(16)));
typedef __attribute__((address_space(1))) v4i* gv4p;  // global int4*
typedef __attribute__((address_space(3))) v4i* lv4p;  // LDS int4*

#define FIN    128
#define FOUT   64
#define BETAc  0.5f
#define ALPHAc 0.2f

// ---------------------------------------------------------------------------
// Kernel 0: pack B = [W | W@a1 | W@a2 | 0...] (128 x 80) into f16 WMMA
// B-fragment layout: 20 fragments (kk=0..3 K-tiles, nt=0..4 N-tiles),
// each fragment = 32 lanes x 16 halves so a wave loads its v16h contiguously.
// Appending W@a[:64] and W@a[64:] as columns folds s1/s2 into the GEMM.
// ---------------------------------------------------------------------------
__global__ __launch_bounds__(256) void gat_prep_kernel(
    const float* __restrict__ W, const float* __restrict__ a,
    _Float16* __restrict__ Bpack) {
  __shared__ float lwa[2][FIN];
  const int tid = threadIdx.x;
  if (tid < FIN) {
    float w1 = 0.f, w2 = 0.f;
    const float* wr = W + tid * FOUT;
#pragma unroll 8
    for (int f = 0; f < FOUT; ++f) {
      w1 += wr[f] * a[f];
      w2 += wr[f] * a[FOUT + f];
    }
    lwa[0][tid] = w1;
    lwa[1][tid] = w2;
  }
  __syncthreads();
  for (int idx = tid; idx < 20 * 512; idx += 256) {
    const int j    = idx & 15;
    const int lane = (idx >> 4) & 31;
    const int frag = idx >> 9;
    const int kk = frag / 5, nt = frag % 5;
    const int k = kk * 32 + ((lane >= 16) ? 8 : 0) + ((j < 8) ? j : (j + 8));
    const int n = nt * 16 + (lane & 15);
    float v;
    if (n < FOUT)            v = W[k * FOUT + n];
    else if (n == FOUT)      v = lwa[0][k];
    else if (n == FOUT + 1)  v = lwa[1][k];
    else                     v = 0.f;
    Bpack[idx] = (_Float16)v;
  }
}

// ---------------------------------------------------------------------------
// Zero an int buffer (per-node degree counters). Runs every launch.
// ---------------------------------------------------------------------------
__global__ void gat_zero_kernel(int* __restrict__ p, int n) {
  const int stride = gridDim.x * blockDim.x;
  for (int i = blockIdx.x * blockDim.x + threadIdx.x; i < n; i += stride)
    p[i] = 0;
}

// ---------------------------------------------------------------------------
// Kernel A: h = input @ [W|Wa1|Wa2] via v_wmma_f32_16x16x32_f16.
// One wave per 16-row node tile; 4 waves/block; B staged in LDS via the
// CDNA5 async global->LDS path (ASYNCcnt) when the toolchain exposes it.
// ---------------------------------------------------------------------------
__global__ __launch_bounds__(128) void gat_gemm_kernel(
    const float* __restrict__ input, const _Float16* __restrict__ Bpack,
    float* __restrict__ h, float* __restrict__ s1, float* __restrict__ s2,
    int nTiles) {
  __shared__ __align__(32) _Float16 lB[20 * 512];
#if __has_builtin(__builtin_amdgcn_global_load_async_to_lds_b128) && \
    __has_builtin(__builtin_amdgcn_s_wait_asynccnt)
  {
    char* g = (char*)Bpack;  // constness irrelevant for the builtin signature
    char* l = (char*)lB;
    for (int i = threadIdx.x; i < 1280; i += 128) {
      __builtin_amdgcn_global_load_async_to_lds_b128(
          (gv4p)(g + i * 16), (lv4p)(l + i * 16), 0, 0);
    }
    __builtin_amdgcn_s_wait_asynccnt(0);
  }
#else
  {
    const float4* src = (const float4*)Bpack;
    float4*       dst = (float4*)lB;
    for (int i = threadIdx.x; i < 1280; i += 128) dst[i] = src[i];
  }
#endif
  __syncthreads();

  const int t = blockIdx.x * 4 + (threadIdx.x >> 5);
  if (t >= nTiles) return;
  const int lane = threadIdx.x & 31;
  const int m0   = t * 16;
  const float* arow = input + (size_t)(m0 + (lane & 15)) * FIN;
  const int o = (lane >= 16) ? 8 : 0;

  v8f acc[5];
#pragma unroll
  for (int nt = 0; nt < 5; ++nt)
    acc[nt] = (v8f){0.f, 0.f, 0.f, 0.f, 0.f, 0.f, 0.f, 0.f};

#pragma unroll
  for (int kk = 0; kk < 4; ++kk) {
    const int kb = kk * 32 + o;
    const float4 f0 = *(const float4*)(arow + kb);
    const float4 f1 = *(const float4*)(arow + kb + 4);
    const float4 f2 = *(const float4*)(arow + kb + 16);
    const float4 f3 = *(const float4*)(arow + kb + 20);
    v16h af;
    af[0]  = (_Float16)f0.x; af[1]  = (_Float16)f0.y;
    af[2]  = (_Float16)f0.z; af[3]  = (_Float16)f0.w;
    af[4]  = (_Float16)f1.x; af[5]  = (_Float16)f1.y;
    af[6]  = (_Float16)f1.z; af[7]  = (_Float16)f1.w;
    af[8]  = (_Float16)f2.x; af[9]  = (_Float16)f2.y;
    af[10] = (_Float16)f2.z; af[11] = (_Float16)f2.w;
    af[12] = (_Float16)f3.x; af[13] = (_Float16)f3.y;
    af[14] = (_Float16)f3.z; af[15] = (_Float16)f3.w;
#pragma unroll
    for (int nt = 0; nt < 5; ++nt) {
      const v16h b = *(const v16h*)&lB[(kk * 5 + nt) * 512 + lane * 16];
      acc[nt] = __builtin_amdgcn_wmma_f32_16x16x32_f16(
          false, af, false, b, (short)0, acc[nt], false, false);
    }
  }

  // C/D layout: lanes 0-15 rows M=0..7, lanes 16-31 rows M=8..15.
  const int rbase = m0 + ((lane >> 4) << 3);
  const int c     = lane & 15;
#pragma unroll
  for (int nt = 0; nt < 4; ++nt) {
#pragma unroll
    for (int i = 0; i < 8; ++i)
      h[(size_t)(rbase + i) * FOUT + nt * 16 + c] = acc[nt][i];
  }
  if (c == 0) {
#pragma unroll
    for (int i = 0; i < 8; ++i) s1[rbase + i] = acc[4][i];
  } else if (c == 1) {
#pragma unroll
    for (int i = 0; i < 8; ++i) s2[rbase + i] = acc[4][i];
  }
}

// ---------------------------------------------------------------------------
// Kernel B1: per-node out-degree histogram (integer atomics, 1 per edge).
// ---------------------------------------------------------------------------
__global__ __launch_bounds__(256) void gat_count_kernel(
    const int* __restrict__ edge, int* __restrict__ counts, int E) {
  const int stride = gridDim.x * blockDim.x;
  for (int i = blockIdx.x * blockDim.x + threadIdx.x; i < E; i += stride)
    atomicAdd(&counts[edge[i]], 1);
}

// ---------------------------------------------------------------------------
// Kernel B2: single-block exclusive scan of counts -> offsets (+cursor copy).
// 1024 threads, each owns a contiguous chunk; Hillis-Steele scan of partials.
// ---------------------------------------------------------------------------
__global__ __launch_bounds__(1024) void gat_scan_kernel(
    const int* __restrict__ counts, int* __restrict__ offsets,
    int* __restrict__ cursor, int N, int E) {
  __shared__ int part[1024];
  const int t     = threadIdx.x;
  const int chunk = (N + 1023) / 1024;
  const int beg   = t * chunk;
  const int end   = (beg + chunk < N) ? (beg + chunk) : N;
  int s = 0;
  for (int i = beg; i < end; ++i) s += counts[i];
  part[t] = s;
  __syncthreads();
#pragma unroll
  for (int d = 1; d < 1024; d <<= 1) {
    const int v = (t >= d) ? part[t - d] : 0;
    __syncthreads();
    part[t] += v;
    __syncthreads();
  }
  int run = (t == 0) ? 0 : part[t - 1];
  for (int i = beg; i < end; ++i) {
    offsets[i] = run;
    cursor[i]  = run;
    run += counts[i];
  }
  if (t == 1023) offsets[N] = E;
}

// ---------------------------------------------------------------------------
// Kernel B3: per-edge attention + CSR scatter. Stores (dst, att) packed so
// the aggregation pass never touches the raw edge list again.
// ---------------------------------------------------------------------------
__global__ __launch_bounds__(256) void gat_scatter_kernel(
    const int* __restrict__ edge, const float* __restrict__ Mv,
    const float* __restrict__ s1, const float* __restrict__ s2,
    int* __restrict__ cursor, int2* __restrict__ csr, int E) {
  const int stride = gridDim.x * blockDim.x;
  for (int i = blockIdx.x * blockDim.x + threadIdx.x; i < E; i += stride) {
    const int src = edge[i];
    const int dst = edge[E + i];
    const float bias = BETAc * Mv[i] + (1.f - BETAc);
    const float z    = s1[src] + bias * s2[dst];
    const float att  = __expf(z > 0.f ? z : ALPHAc * z);
    const int pos = atomicAdd(&cursor[src], 1);
    csr[pos] = make_int2(dst, __float_as_int(att));
  }
}

// ---------------------------------------------------------------------------
// Kernel B4: wave-per-node aggregation. Pure register accumulation — no FP
// atomics. Each lane owns 2 features (float2 gather of h[dst] = 256 B/edge,
// L2-resident). Normalize + ELU fused into the epilogue.
// ---------------------------------------------------------------------------
__global__ __launch_bounds__(256) void gat_aggregate_kernel(
    const int* __restrict__ offsets, const int2* __restrict__ csr,
    const float* __restrict__ h, float* __restrict__ out, int N) {
  const int u = (blockIdx.x * blockDim.x + threadIdx.x) >> 5;
  if (u >= N) return;
  const int lane = threadIdx.x & 31;
  const int beg = offsets[u];
  const int end = offsets[u + 1];
  float a0 = 0.f, a1 = 0.f, rs = 0.f;
  for (int j = beg; j < end; ++j) {
    const int2  c   = csr[j];
    const float att = __int_as_float(c.y);
    const float2 hv = *(const float2*)(h + (size_t)c.x * FOUT + lane * 2);
    a0 += att * hv.x;
    a1 += att * hv.y;
    rs += att;
  }
  const float inv = 1.f / rs;
  float x0 = a0 * inv;
  float x1 = a1 * inv;
  x0 = (x0 > 0.f) ? x0 : (__expf(x0) - 1.f);
  x1 = (x1 > 0.f) ? x1 : (__expf(x1) - 1.f);
  *(float2*)(out + (size_t)u * FOUT + lane * 2) = make_float2(x0, x1);
}

// ---------------------------------------------------------------------------
extern "C" void kernel_launch(void* const* d_in, const int* in_sizes, int n_in,
                              void* d_out, int out_size, void* d_ws, size_t ws_size,
                              hipStream_t stream) {
  const float* input = (const float*)d_in[0];  // [N,128]
  const float* Mv    = (const float*)d_in[1];  // [E]
  const float* W     = (const float*)d_in[2];  // [128,64]
  const float* a     = (const float*)d_in[3];  // [128]
  const int*   edge  = (const int*)d_in[4];    // [2,E]

  const int N = in_sizes[0] / FIN;
  const int E = in_sizes[4] / 2;

  char* ws = (char*)d_ws;
  size_t off = 0;
  auto alloc = [&](size_t bytes) {
    size_t o = off;
    off = (off + bytes + 15) & ~(size_t)15;
    return o;
  };
  float*    hF32    = (float*)(ws + alloc((size_t)N * FOUT * 4));
  float*    s1      = (float*)(ws + alloc((size_t)N * 4));
  float*    s2      = (float*)(ws + alloc((size_t)N * 4));
  int*      counts  = (int*)(ws + alloc((size_t)N * 4));
  int*      offsets = (int*)(ws + alloc((size_t)(N + 1) * 4));
  int*      cursor  = (int*)(ws + alloc((size_t)N * 4));
  _Float16* Bpack   = (_Float16*)(ws + alloc((size_t)20 * 512 * 2));
  int2*     csr     = (int2*)(ws + alloc((size_t)E * 8));

  float* out = (float*)d_out;

  // 0) pack W (+ appended W@a columns) into f16 B fragments
  gat_prep_kernel<<<1, 256, 0, stream>>>(W, a, Bpack);

  // 1) zero degree counters (every launch -> replay-safe)
  gat_zero_kernel<<<(N + 255) / 256, 256, 0, stream>>>(counts, N);

  // 2) WMMA GEMM producing h, s1, s2
  const int nTiles = (N + 15) / 16;
  gat_gemm_kernel<<<(nTiles + 3) / 4, 128, 0, stream>>>(input, Bpack, hF32, s1, s2, nTiles);

  // 3) CSR build: histogram -> scan -> attention scatter
  gat_count_kernel<<<2048, 256, 0, stream>>>(edge, counts, E);
  gat_scan_kernel<<<1, 1024, 0, stream>>>(counts, offsets, cursor, N, E);
  gat_scatter_kernel<<<2048, 256, 0, stream>>>(edge, Mv, s1, s2, cursor, csr, E);

  // 4) wave-per-node aggregation + normalize + ELU (no FP atomics)
  gat_aggregate_kernel<<<(N * 32 + 255) / 256, 256, 0, stream>>>(offsets, csr, hF32, out, N);
}
